// FCSSM_31593779429661
// MI455X (gfx1250) — compile-verified, matrix operand order checked by reading, not run on hardware
//
#include <hip/hip_runtime.h>
#include <hip/hip_bf16.h>

// ---------------------------------------------------------------------------
// FCSSM forward for MI455X (gfx1250, wave32, WMMA).
//   GEMMs -> v_wmma_f32_16x16x32_bf16, 32x64 tile / wave (8 wmma per K-step).
//   Selective scan -> register state h[16]/thread, async-to-LDS double-buffered
//   B(t)/C(t) staging (global_load_async_to_lds_b128 + s_wait_asynccnt).
// ---------------------------------------------------------------------------

#define BATCH   2
#define SEQLEN  1024
#define DMODEL  1024
#define DINNER  2048
#define DSTATE  16
#define DTRANK  64
#define DCONV   4
#define NTOK    (BATCH * SEQLEN)          // 2048 tokens
#define XDBL    (DTRANK + 2 * DSTATE)     // 96

typedef __attribute__((ext_vector_type(16))) __bf16 v16bf;
typedef __attribute__((ext_vector_type(8)))  __bf16 v8bf;
typedef __attribute__((ext_vector_type(8)))  float  v8f;

static __device__ __forceinline__ __bf16 f2bf(float f) {
    __hip_bfloat16 h = __float2bfloat16(f);
    return *reinterpret_cast<__bf16*>(&h);
}

// Fragment load per ISA 7.12.2 (16-bit A 16x32):
//   lanes 0-15:  elems 0-7 -> K=k0..k0+7,    elems 8-15 -> K=k0+16..k0+23
//   lanes 16-31: elems 0-7 -> K=k0+8..k0+15, elems 8-15 -> K=k0+24..k0+31
// p already includes row*ld + half*8; two contiguous b128 loads.
static __device__ __forceinline__ v16bf ldfrag(const __bf16* p) {
    v8bf lo = *(const v8bf*)p;
    v8bf hi = *(const v8bf*)(p + 16);
    return __builtin_shufflevector(lo, hi,
               0,1,2,3,4,5,6,7,8,9,10,11,12,13,14,15);
}

// ------------------------------ fp32 -> bf16 -------------------------------
__global__ void cvt_bf16_kernel(const float* __restrict__ in,
                                __bf16* __restrict__ out, int n) {
    int i = blockIdx.x * blockDim.x + threadIdx.x;
    if (i < n) out[i] = f2bf(in[i]);
}

// ---------------------- WMMA GEMM, 32x64 tile per wave ---------------------
// C[M,N] = A[M,K](bf16,lda) @ W[N,K]^T(bf16,ldb), C fp32(ldc).
// Requires M%32==0, N%64==0, K%32==0. 2 A-frags x 4 B-frags -> 8 wmma / step.
__global__ void wmma_gemm_bt_2x4_kernel(const __bf16* __restrict__ A,
                                        const __bf16* __restrict__ W,
                                        float* __restrict__ C,
                                        int M, int N, int K,
                                        int lda, int ldb, int ldc) {
    const int wave  = blockIdx.x * (blockDim.x >> 5) + (threadIdx.x >> 5);
    const int nBig  = N >> 6;
    const int total = (M >> 5) * nBig;
    if (wave >= total) return;
    const int mt = wave / nBig;
    const int nt = wave % nBig;

    const int lane = threadIdx.x & 31;
    const int half = lane >> 4;
    const int r    = lane & 15;

    const __bf16* a0 = A + (size_t)(mt * 32 + r) * lda + half * 8;
    const __bf16* a1 = a0 + (size_t)16 * lda;
    const __bf16* b0 = W + (size_t)(nt * 64 + r) * ldb + half * 8;

    v8f acc[8] = {};
    for (int k0 = 0; k0 < K; k0 += 32) {
        __builtin_prefetch(a0 + k0 + 32, 0, 3);
        __builtin_prefetch(b0 + k0 + 32, 0, 3);
        v16bf av0 = ldfrag(a0 + k0);
        v16bf av1 = ldfrag(a1 + k0);
        #pragma unroll
        for (int j = 0; j < 4; ++j) {
            v16bf bv = ldfrag(b0 + (size_t)j * 16 * ldb + k0);
            acc[j]     = __builtin_amdgcn_wmma_f32_16x16x32_bf16(
                             false, av0, false, bv, (short)0, acc[j],     false, false);
            acc[4 + j] = __builtin_amdgcn_wmma_f32_16x16x32_bf16(
                             false, av1, false, bv, (short)0, acc[4 + j], false, false);
        }
    }
    // C/D layout: lane -> n = lane&15 ; VGPR j -> m = half*8 + j
    #pragma unroll
    for (int t = 0; t < 2; ++t) {
        #pragma unroll
        for (int j = 0; j < 4; ++j) {
            float* crow = C + (size_t)(mt * 32 + t * 16 + half * 8) * ldc
                            + nt * 64 + j * 16 + r;
            v8f c = acc[t * 4 + j];
            #pragma unroll
            for (int jj = 0; jj < 8; ++jj) crow[(size_t)jj * ldc] = c[jj];
        }
    }
}

// ------------------ WMMA GEMM, 16x16 tile per wave (small N) ---------------
__global__ void wmma_gemm_bt_kernel(const __bf16* __restrict__ A,
                                    const __bf16* __restrict__ W,
                                    float* __restrict__ C,
                                    int M, int N, int K,
                                    int lda, int ldb, int ldc) {
    const int tile   = blockIdx.x * (blockDim.x >> 5) + (threadIdx.x >> 5);
    const int ntN    = N >> 4;
    const int ntiles = (M >> 4) * ntN;
    if (tile >= ntiles) return;
    const int mt = tile / ntN;
    const int nt = tile % ntN;
    const int lane = threadIdx.x & 31;
    const int half = lane >> 4;
    const int r    = lane & 15;

    const __bf16* arow = A + (size_t)(mt * 16 + r) * lda + half * 8;
    const __bf16* brow = W + (size_t)(nt * 16 + r) * ldb + half * 8;

    v8f acc = {};
    for (int k0 = 0; k0 < K; k0 += 32) {
        v16bf av = ldfrag(arow + k0);
        v16bf bv = ldfrag(brow + k0);
        acc = __builtin_amdgcn_wmma_f32_16x16x32_bf16(
                  false, av, false, bv, (short)0, acc, false, false);
    }
    float* crow = C + (size_t)(mt * 16 + half * 8) * ldc + nt * 16 + r;
    #pragma unroll
    for (int j = 0; j < 8; ++j) crow[(size_t)j * ldc] = acc[j];
}

// --------------------- depthwise causal conv1d + SiLU ----------------------
__global__ void conv_silu_kernel(const float* __restrict__ xz,       // [B,L,2*Di]
                                 const float* __restrict__ conv_w,   // [Di,1,4]
                                 const float* __restrict__ conv_b,   // [Di]
                                 float*  __restrict__ xc,            // [B,L,Di]
                                 __bf16* __restrict__ xc_bf) {
    int idx = blockIdx.x * blockDim.x + threadIdx.x;
    if (idx >= NTOK * DINNER) return;
    int d   = idx % DINNER;
    int tok = idx / DINNER;
    int l   = tok % SEQLEN;
    float acc = conv_b[d];
    #pragma unroll
    for (int j = 0; j < DCONV; ++j) {
        int lp = l - (DCONV - 1) + j;
        if (lp >= 0) {
            float v = xz[(size_t)(tok + (lp - l)) * (2 * DINNER) + d];
            acc = fmaf(conv_w[d * DCONV + j], v, acc);
        }
    }
    float s = acc / (1.0f + __expf(-acc));   // SiLU
    xc[idx]    = s;
    xc_bf[idx] = f2bf(s);
}

// ----------------- softplus(dt) * exp(clip(FiLM log-gain)) -----------------
__global__ void delta_gain_kernel(float* __restrict__ delta,
                                  const float* __restrict__ dt_proj_b,
                                  const int*   __restrict__ band_idx,
                                  const float* __restrict__ mask_frac,
                                  const float* __restrict__ rho_bar,
                                  const float* __restrict__ s_band_dt,
                                  const float* __restrict__ s_rho_w,
                                  const float* __restrict__ s_mask_w) {
    int idx = blockIdx.x * blockDim.x + threadIdx.x;
    if (idx >= NTOK * DINNER) return;
    int d   = idx % DINNER;
    int tok = idx / DINNER;
    int l   = tok % SEQLEN;
    float x  = delta[idx] + dt_proj_b[d];
    float sp = (x > 20.0f) ? x : log1pf(__expf(x));
    float g  = s_band_dt[band_idx[l] * DINNER + d]
             + mask_frac[tok] * s_mask_w[d]
             + rho_bar[l] * s_rho_w[d];
    g = fminf(fmaxf(g, -2.0f), 2.0f);
    delta[idx] = sp * __expf(g);
}

// ----------------------- B / C token FiLM gains ----------------------------
__global__ void bc_gain_kernel(const float* __restrict__ x_dbl,
                               const int*   __restrict__ band_idx,
                               const float* __restrict__ mask_frac,
                               const float* __restrict__ rho_bar,
                               const float* __restrict__ s_band_B,
                               const float* __restrict__ s_rho_B,
                               const float* __restrict__ s_mask_B,
                               const float* __restrict__ s_band_C,
                               const float* __restrict__ s_rho_C,
                               const float* __restrict__ s_mask_C,
                               float* __restrict__ Btok,
                               float* __restrict__ Ctok) {
    int idx = blockIdx.x * blockDim.x + threadIdx.x;
    if (idx >= NTOK * DSTATE) return;
    int n   = idx % DSTATE;
    int tok = idx / DSTATE;
    int l   = tok % SEQLEN;
    int bb  = band_idx[l];
    float mf = mask_frac[tok], rb = rho_bar[l];

    float gB = s_band_B[bb * DSTATE + n] + mf * s_mask_B[n] + rb * s_rho_B[n];
    gB = fminf(fmaxf(gB, -2.0f), 2.0f);
    Btok[idx] = x_dbl[(size_t)tok * XDBL + DTRANK + n] * __expf(gB);

    float gC = s_band_C[bb * DSTATE + n] + mf * s_mask_C[n] + rb * s_rho_C[n];
    gC = fminf(fmaxf(gC, -2.0f), 2.0f);
    Ctok[idx] = x_dbl[(size_t)tok * XDBL + DTRANK + DSTATE + n] * __expf(gC);
}

// --------------------------- selective scan --------------------------------
// One thread per (b, d): 16-state recurrence entirely in VGPRs.
// Per-timestep B(t)/C(t) (128 B, shared across all d) are double-buffered in
// LDS via CDNA5 async-to-LDS copies issued by lanes 0-7 of wave 0, tracked
// with ASYNCcnt (s_wait_asynccnt) and published with the workgroup barrier.
// The next step's fetch overlaps the current step's exp/fma chain.
__global__ void scan_kernel(const float* __restrict__ delta,      // [B,L,Di]
                            const float* __restrict__ xc,         // [B,L,Di]
                            const float* __restrict__ Btok,       // [B,L,16]
                            const float* __restrict__ Ctok,       // [B,L,16]
                            const float* __restrict__ lambda_log, // [Di,16]
                            float* __restrict__ ys)               // [B,L,Di]
{
    __shared__ float sBC[2][2 * DSTATE];   // [buf][ B(16) | C(16) ]

    const int b = blockIdx.x / (DINNER / 256);
    const int d = (blockIdx.x % (DINNER / 256)) * 256 + threadIdx.x;

    float lam[DSTATE], h[DSTATE];
    #pragma unroll
    for (int n = 0; n < DSTATE; ++n) {
        lam[n] = __expf(lambda_log[d * DSTATE + n]);
        h[n]   = 0.0f;
    }

    // lanes 0-7: async-copy 16B each (4 chunks of B, 4 chunks of C) into buf.
    auto fetch = [&](int tok, int buf) {
        const int i     = threadIdx.x;          // < 8
        const int sel   = i >> 2;               // 0 = B, 1 = C
        const int chunk = i & 3;
        const float* src = (sel ? Ctok : Btok) + (size_t)tok * DSTATE + chunk * 4;
        unsigned long long gaddr = (unsigned long long)(uintptr_t)src;
        unsigned int loff = (unsigned int)(uintptr_t)&sBC[buf][sel * DSTATE + chunk * 4];
        asm volatile("global_load_async_to_lds_b128 %0, %1, off"
                     :: "v"(loff), "v"(gaddr) : "memory");
    };

    if (threadIdx.x < 8) fetch(b * SEQLEN, 0);

    for (int l = 0; l < SEQLEN; ++l) {
        const int tok = b * SEQLEN + l;
        const int cur = l & 1;
        if (threadIdx.x < 8) {
            asm volatile("s_wait_asynccnt 0x0" ::: "memory");
        }
        __syncthreads();                         // publish buf `cur` to all waves
        if (l + 1 < SEQLEN && threadIdx.x < 8) fetch(tok + 1, cur ^ 1);

        const float* sB = &sBC[cur][0];
        const float* sC = &sBC[cur][DSTATE];
        const float dt = delta[(size_t)tok * DINNER + d];
        const float u  = xc[(size_t)tok * DINNER + d];
        float y = 0.0f;
        #pragma unroll
        for (int n = 0; n < DSTATE; ++n) {
            float dl = dt * lam[n];
            float a  = __expf(-dl);
            float ratio = (fabsf(dl) < 1e-4f) ? (1.0f - 0.5f * dl)
                                              : ((1.0f - a) / fmaxf(dl, 1e-8f));
            h[n] = fmaf(a, h[n], dt * sB[n] * ratio * u);
            y    = fmaf(h[n], sC[n], y);
        }
        ys[(size_t)tok * DINNER + d] = y;
    }
}

// ------------------------- skip + gate (SiLU(z)) ---------------------------
__global__ void gate_kernel(const float* __restrict__ ys,
                            const float* __restrict__ xc,
                            const float* __restrict__ xz,
                            const float* __restrict__ D_param,
                            __bf16* __restrict__ gated_bf) {
    int idx = blockIdx.x * blockDim.x + threadIdx.x;
    if (idx >= NTOK * DINNER) return;
    int d   = idx % DINNER;
    int tok = idx / DINNER;
    float z  = xz[(size_t)tok * (2 * DINNER) + DINNER + d];
    float sz = z / (1.0f + __expf(-z));
    float o  = (ys[idx] + xc[idx] * D_param[d]) * sz;
    gated_bf[idx] = f2bf(o);
}

// ---------------------------------------------------------------------------
static inline size_t align256(size_t x) { return (x + 255) & ~(size_t)255; }
static inline int cdiv(int a, int b) { return (a + b - 1) / b; }

extern "C" void kernel_launch(void* const* d_in, const int* in_sizes, int n_in,
                              void* d_out, int out_size, void* d_ws, size_t ws_size,
                              hipStream_t stream) {
    const float* x          = (const float*)d_in[0];
    const int*   band_idx   = (const int*)  d_in[1];
    const float* mask_frac  = (const float*)d_in[2];
    const float* rho_bar    = (const float*)d_in[3];
    const float* in_proj_w  = (const float*)d_in[4];
    const float* conv_w     = (const float*)d_in[5];
    const float* conv_b     = (const float*)d_in[6];
    const float* x_proj_w   = (const float*)d_in[7];
    const float* dt_proj_w  = (const float*)d_in[8];
    const float* dt_proj_b  = (const float*)d_in[9];
    const float* lambda_log = (const float*)d_in[10];
    const float* D_param    = (const float*)d_in[11];
    const float* out_proj_w = (const float*)d_in[12];
    const float* s_band_dt  = (const float*)d_in[13];
    const float* s_rho_dt_w = (const float*)d_in[14];
    const float* s_mask_dt_w= (const float*)d_in[15];
    const float* s_band_B   = (const float*)d_in[16];
    const float* s_rho_B_w  = (const float*)d_in[17];
    const float* s_mask_B_w = (const float*)d_in[18];
    const float* s_band_C   = (const float*)d_in[19];
    const float* s_rho_C_w  = (const float*)d_in[20];
    const float* s_mask_C_w = (const float*)d_in[21];

    // -------- workspace layout (all 256B aligned) --------
    char* ws = (char*)d_ws;
    size_t off = 0;
    auto carve = [&](size_t bytes) { char* p = ws + off; off += align256(bytes); return p; };
    __bf16* x_bf     = (__bf16*)carve((size_t)NTOK * DMODEL * 2);
    __bf16* w_in_bf  = (__bf16*)carve((size_t)2 * DINNER * DMODEL * 2);
    __bf16* w_xp_bf  = (__bf16*)carve((size_t)XDBL * DINNER * 2);
    __bf16* w_dt_bf  = (__bf16*)carve((size_t)DINNER * DTRANK * 2);
    __bf16* w_out_bf = (__bf16*)carve((size_t)DMODEL * DINNER * 2);
    float*  xz       = (float*) carve((size_t)NTOK * 2 * DINNER * 4);
    float*  xc       = (float*) carve((size_t)NTOK * DINNER * 4);
    __bf16* xc_bf    = (__bf16*)carve((size_t)NTOK * DINNER * 2);
    float*  x_dbl    = (float*) carve((size_t)NTOK * XDBL * 4);
    __bf16* x_dbl_bf = (__bf16*)carve((size_t)NTOK * XDBL * 2);
    float*  delta    = (float*) carve((size_t)NTOK * DINNER * 4);
    float*  Btok     = (float*) carve((size_t)NTOK * DSTATE * 4);
    float*  Ctok     = (float*) carve((size_t)NTOK * DSTATE * 4);
    float*  ys       = (float*) carve((size_t)NTOK * DINNER * 4);
    __bf16* gated_bf = (__bf16*)carve((size_t)NTOK * DINNER * 2);
    (void)ws_size; (void)in_sizes; (void)n_in; (void)out_size;

    const int T = 256;
    auto cvt = [&](const float* src, __bf16* dst, int n) {
        cvt_bf16_kernel<<<cdiv(n, T), T, 0, stream>>>(src, dst, n);
    };
    // big-tile GEMM (M%32, N%64, K%32)
    auto gemm = [&](const __bf16* A, const __bf16* W, float* C,
                    int M, int N, int K, int lda, int ldb, int ldc) {
        int waves = (M / 32) * (N / 64);
        wmma_gemm_bt_2x4_kernel<<<cdiv(waves, 8), 256, 0, stream>>>(
            A, W, C, M, N, K, lda, ldb, ldc);
    };
    auto gemm_small = [&](const __bf16* A, const __bf16* W, float* C,
                          int M, int N, int K, int lda, int ldb, int ldc) {
        int tiles = (M / 16) * (N / 16);
        wmma_gemm_bt_kernel<<<cdiv(tiles, 8), 256, 0, stream>>>(
            A, W, C, M, N, K, lda, ldb, ldc);
    };

    // 0) precision conversions
    cvt(x,          x_bf,     NTOK * DMODEL);
    cvt(in_proj_w,  w_in_bf,  2 * DINNER * DMODEL);
    cvt(x_proj_w,   w_xp_bf,  XDBL * DINNER);
    cvt(dt_proj_w,  w_dt_bf,  DINNER * DTRANK);
    cvt(out_proj_w, w_out_bf, DMODEL * DINNER);

    // 1) in_proj: xz[NTOK, 4096] = x @ in_proj_w^T
    gemm(x_bf, w_in_bf, xz, NTOK, 2 * DINNER, DMODEL, DMODEL, DMODEL, 2 * DINNER);

    // 2) depthwise causal conv + SiLU -> xc (f32 + bf16)
    conv_silu_kernel<<<cdiv(NTOK * DINNER, T), T, 0, stream>>>(
        xz, conv_w, conv_b, xc, xc_bf);

    // 3) x_proj: x_dbl[NTOK, 96] = xc @ x_proj_w^T  (N=96 -> 16x16 kernel)
    gemm_small(xc_bf, w_xp_bf, x_dbl, NTOK, XDBL, DINNER, DINNER, DINNER, XDBL);
    cvt(x_dbl, x_dbl_bf, NTOK * XDBL);

    // 4) dt_proj: delta_pre[NTOK, DINNER] = x_dbl[:, :64] @ dt_proj_w^T
    gemm(x_dbl_bf, w_dt_bf, delta, NTOK, DINNER, DTRANK, XDBL, DTRANK, DINNER);

    // 5) softplus + FiLM gain on delta (in place)
    delta_gain_kernel<<<cdiv(NTOK * DINNER, T), T, 0, stream>>>(
        delta, dt_proj_b, band_idx, mask_frac, rho_bar,
        s_band_dt, s_rho_dt_w, s_mask_dt_w);

    // 6) B / C token gains
    bc_gain_kernel<<<cdiv(NTOK * DSTATE, T), T, 0, stream>>>(
        x_dbl, band_idx, mask_frac, rho_bar,
        s_band_B, s_rho_B_w, s_mask_B_w,
        s_band_C, s_rho_C_w, s_mask_C_w, Btok, Ctok);

    // 7) selective scan (register-resident state, async LDS double buffering)
    scan_kernel<<<BATCH * (DINNER / 256), 256, 0, stream>>>(
        delta, xc, Btok, Ctok, lambda_log, ys);

    // 8) skip + gate -> bf16
    gate_kernel<<<cdiv(NTOK * DINNER, T), T, 0, stream>>>(
        ys, xc, xz, D_param, gated_bf);

    // 9) out_proj: out[NTOK, 1024] = gated @ out_proj_w^T -> d_out (fp32)
    gemm(gated_bf, w_out_bf, (float*)d_out, NTOK, DMODEL, DINNER, DINNER, DINNER, DMODEL);
}
